// temporal_cross_attention_6811818131528
// MI455X (gfx1250) — matrix operand order repaired and checked by reading, not compiled, not run on hardware
//
#include <hip/hip_runtime.h>
#include <hip/hip_bf16.h>

// ---- problem constants (match reference) ----
#define Bsz 2
#define Tq  2048
#define Nk  2048
#define Dd  1024
#define Ll  1024
#define Hh  16
#define HD  64

typedef _Float16 v16h __attribute__((ext_vector_type(16)));
typedef _Float16 v8h  __attribute__((ext_vector_type(8)));
typedef float    v8f  __attribute__((ext_vector_type(8)));

union V16 { v16h v; v8h h[2]; };

// ---------------------------------------------------------------------------
// LayerNorm (f32 in) -> f16 out.  One row per block, blockDim = 256, C = 1024.
// ---------------------------------------------------------------------------
__global__ __launch_bounds__(256) void ln_f16_kernel(
    const float* __restrict__ in, const float* __restrict__ gamma,
    const float* __restrict__ beta, _Float16* __restrict__ out, int C)
{
    __shared__ float red[2][8];
    const int row = blockIdx.x;
    const float* p = in + (size_t)row * C;
    float s = 0.f, ss = 0.f;
    for (int i = threadIdx.x; i < C; i += blockDim.x) {
        float v = p[i]; s += v; ss += v * v;
    }
    for (int off = 16; off > 0; off >>= 1) {
        s  += __shfl_xor(s,  off, 32);
        ss += __shfl_xor(ss, off, 32);
    }
    const int wid = threadIdx.x >> 5, lane = threadIdx.x & 31;
    if (lane == 0) { red[0][wid] = s; red[1][wid] = ss; }
    __syncthreads();
    if (threadIdx.x == 0) {
        float a = 0.f, b = 0.f;
        const int nw = blockDim.x >> 5;
        for (int w = 0; w < nw; ++w) { a += red[0][w]; b += red[1][w]; }
        red[0][0] = a; red[1][0] = b;
    }
    __syncthreads();
    const float mean = red[0][0] / (float)C;
    const float var  = red[1][0] / (float)C - mean * mean;
    const float rstd = rsqrtf(var + 1e-5f);
    for (int i = threadIdx.x; i < C; i += blockDim.x) {
        float v = (p[i] - mean) * rstd * gamma[i] + beta[i];
        out[(size_t)row * C + i] = (_Float16)v;
    }
}

// ---------------------------------------------------------------------------
// Weight transpose + f32->f16 convert:  Wt[n][k] = (f16) W[k][n]
// ---------------------------------------------------------------------------
__global__ __launch_bounds__(256) void wt_f16_kernel(
    const float* __restrict__ W, _Float16* __restrict__ Wt, int K, int Ncol)
{
    int idx = blockIdx.x * blockDim.x + threadIdx.x;
    if (idx >= K * Ncol) return;
    int k = idx / Ncol, n = idx % Ncol;
    Wt[(size_t)n * K + k] = (_Float16)W[idx];
}

// ---------------------------------------------------------------------------
// WMMA GEMM:  C(M x Ncol) = A(M x K) * Bt(Ncol x K)^T, f16 in, f32 acc.
// One 16x64 tile per wave (4 accumulators share one A operand per K-step),
// 4 waves/block.  K stepped by 32.
// Output layout:
//   mode 0 (q/k): out[((b*16 + h)*Trows + seq)*64 + dd]   (b = grow/Trows)
//   mode 1 (v^T): out[((b*16 + h)*64 + dd)*Trows + seq]
// ---------------------------------------------------------------------------
__global__ __launch_bounds__(128) void gemm_wmma_kernel(
    const _Float16* __restrict__ A, const _Float16* __restrict__ Bt,
    _Float16* __restrict__ out, int M, int K, int Ncol, int mode, int Trows)
{
    const int wid  = threadIdx.x >> 5;
    const int lane = threadIdx.x & 31;
    const int half = lane >> 4;          // 0: lanes 0-15, 1: lanes 16-31
    const int l16  = lane & 15;

    const int tile   = blockIdx.x * 4 + wid;   // over Mtiles * (Ncol/64)
    const int N64    = Ncol >> 6;
    const int mt  = tile / N64;
    const int nt4 = tile % N64;                // 64-column group

    v8f acc[4];
    for (int s = 0; s < 4; ++s) acc[s] = (v8f){};

    const _Float16* arow = A + (size_t)(mt * 16 + l16) * K;   // row m = l16
    const _Float16* brow[4];
    for (int s = 0; s < 4; ++s)                               // col n = l16 of subtile s
        brow[s] = Bt + (size_t)(nt4 * 64 + s * 16 + l16) * K + half * 16;

    for (int k0 = 0; k0 < K; k0 += 32) {
        V16 a;
        a.h[0] = *(const v8h*)(arow + k0 + half * 16);        // K = k0+16*half+0..7
        a.h[1] = *(const v8h*)(arow + k0 + half * 16 + 8);    // K = k0+16*half+8..15
        for (int s = 0; s < 4; ++s) {
            v16h b = *(const v16h*)(brow[s] + k0);            // K = 16*half + e
            acc[s] = __builtin_amdgcn_wmma_f32_16x16x32_f16(
                         false, a.v, false, b, (short)0, acc[s], false, false);
        }
    }

    // C layout: element j -> row j + 8*half, col l16
    for (int s = 0; s < 4; ++s) {
        for (int j = 0; j < 8; ++j) {
            const int row = j + half * 8;
            const int gr  = mt * 16 + row;
            const int gc  = nt4 * 64 + s * 16 + l16;
            const int bb  = gr / Trows, seq = gr % Trows;
            const int h   = gc >> 6,    dd  = gc & 63;
            size_t o;
            if (mode == 0) o = ((size_t)(bb * Hh + h) * Trows + seq) * 64 + dd;
            else           o = ((size_t)(bb * Hh + h) * 64 + dd) * (size_t)Trows + seq;
            out[o] = (_Float16)acc[s][j];
        }
    }
}

// ---------------------------------------------------------------------------
// Flash-style attention.  One wave = 16 query rows of one (b,h).
// Per 32-key step: 4 WMMA for scores, online softmax (shfl row-reductions),
// P staged through LDS into A-layout, 4 WMMA for P*V (hd = 64).
// Epilogue: normalize, add residual, store f32.
// ---------------------------------------------------------------------------
__global__ __launch_bounds__(128) void attn_wmma_kernel(
    const _Float16* __restrict__ q,   // (B,H,T,64)
    const _Float16* __restrict__ k,   // (B,H,N,64)
    const _Float16* __restrict__ vt,  // (B,H,64,N)
    const float* __restrict__ x, float* __restrict__ out)
{
    __shared__ __align__(32) _Float16 Pls[4][16 * 32];

    const int wid  = threadIdx.x >> 5;
    const int lane = threadIdx.x & 31;
    const int half = lane >> 4;
    const int l16  = lane & 15;

    const int tileId = blockIdx.x * 4 + wid;   // over B*H*(T/16)
    const int tTiles = Tq / 16;
    const int t0 = (tileId % tTiles) * 16;
    const int bh = tileId / tTiles;            // b*H + h

    const _Float16* qbase = q  + (size_t)bh * Tq * 64;
    const _Float16* kbase = k  + (size_t)bh * Nk * 64;
    const _Float16* vbase = vt + (size_t)bh * 64 * Nk;

    // Q A-operand: 16x64 as two 16x32 chunks, resident for whole key loop.
    V16 qa[2];
    {
        const _Float16* qr = qbase + (size_t)(t0 + l16) * 64;
        for (int c = 0; c < 2; ++c) {
            qa[c].h[0] = *(const v8h*)(qr + c * 32 + half * 16);
            qa[c].h[1] = *(const v8h*)(qr + c * 32 + half * 16 + 8);
        }
    }

    float mrow[8], lrow[8];
    v8f acc[4];
    for (int j = 0; j < 8; ++j) { mrow[j] = -3.0e38f; lrow[j] = 0.f; }
    for (int d = 0; d < 4; ++d) acc[d] = (v8f){};
    const float scale = 0.125f;                // 1/sqrt(64)
    _Float16* P = &Pls[wid][0];

    for (int n0 = 0; n0 < Nk; n0 += 32) {
        // ---- scores: two 16-key tiles, hd=64 split into two K=32 chunks ----
        v8f s0 = {}, s1 = {};
        for (int c = 0; c < 2; ++c) {
            v16h b0 = *(const v16h*)(kbase + (size_t)(n0 + l16)      * 64 + c * 32 + half * 16);
            v16h b1 = *(const v16h*)(kbase + (size_t)(n0 + 16 + l16) * 64 + c * 32 + half * 16);
            s0 = __builtin_amdgcn_wmma_f32_16x16x32_f16(false, qa[c].v, false, b0, (short)0, s0, false, false);
            s1 = __builtin_amdgcn_wmma_f32_16x16x32_f16(false, qa[c].v, false, b1, (short)0, s1, false, false);
        }

        // ---- online softmax over the 32 keys (rows live per-VGPR in 16-lane halves)
        float p0[8], p1[8];
        for (int j = 0; j < 8; ++j) {
            float a = s0[j] * scale, b = s1[j] * scale;
            float t = fmaxf(a, b);
            for (int off = 1; off < 16; off <<= 1) t = fmaxf(t, __shfl_xor(t, off, 32));
            const float mn = fmaxf(mrow[j], t);
            const float ea = __expf(a - mn), eb = __expf(b - mn);
            float sum = ea + eb;
            for (int off = 1; off < 16; off <<= 1) sum += __shfl_xor(sum, off, 32);
            const float corr = __expf(mrow[j] - mn);
            lrow[j] = lrow[j] * corr + sum;
            mrow[j] = mn;
            for (int d = 0; d < 4; ++d) acc[d][j] *= corr;
            p0[j] = ea; p1[j] = eb;
        }

        // ---- restage P (C layout -> A layout) via per-wave LDS slice ----
        __syncthreads();                       // protect previous iteration's reads
        for (int j = 0; j < 8; ++j) {
            const int row = j + half * 8;
            P[row * 32 + l16]      = (_Float16)p0[j];
            P[row * 32 + 16 + l16] = (_Float16)p1[j];
        }
        __syncthreads();
        V16 pa;
        pa.h[0] = *(const v8h*)(P + l16 * 32 + half * 16);
        pa.h[1] = *(const v8h*)(P + l16 * 32 + half * 16 + 8);

        // ---- P(16x32) * V(32x64): 4 output tiles of 16 cols ----
        for (int d = 0; d < 4; ++d) {
            v16h vb = *(const v16h*)(vbase + (size_t)(d * 16 + l16) * Nk + n0 + half * 16);
            acc[d] = __builtin_amdgcn_wmma_f32_16x16x32_f16(false, pa.v, false, vb, (short)0, acc[d], false, false);
        }
    }

    // ---- epilogue: normalize, residual add, store ----
    const int h  = bh & 15;
    const int bb = bh >> 4;
    for (int j = 0; j < 8; ++j) {
        const int row = j + half * 8;
        const int t   = t0 + row;
        const float inv = 1.f / lrow[j];
        for (int d = 0; d < 4; ++d) {
            const size_t o = ((size_t)(bb * Tq + t)) * Dd + h * 64 + d * 16 + l16;
            out[o] = x[o] + acc[d][j] * inv;
        }
    }
}

// ---------------------------------------------------------------------------
// Launch
// ---------------------------------------------------------------------------
extern "C" void kernel_launch(void* const* d_in, const int* in_sizes, int n_in,
                              void* d_out, int out_size, void* d_ws, size_t ws_size,
                              hipStream_t stream)
{
    const float* x       = (const float*)d_in[0];
    const float* xf      = (const float*)d_in[1];
    const float* Wq      = (const float*)d_in[2];
    const float* Wk      = (const float*)d_in[3];
    const float* Wv      = (const float*)d_in[4];
    const float* gamma_x = (const float*)d_in[5];
    const float* beta_x  = (const float*)d_in[6];
    const float* gamma_f = (const float*)d_in[7];
    const float* beta_f  = (const float*)d_in[8];
    float* out = (float*)d_out;

    char* ws = (char*)d_ws;
    // byte offsets into workspace (total 46 MB)
    _Float16* xn  = (_Float16*)(ws);                        // B*T*D  f16 = 8 MB
    _Float16* fn  = (_Float16*)(ws + ((size_t) 8 << 20));   // B*N*L  f16 = 8 MB
    _Float16* Wqt = (_Float16*)(ws + ((size_t)16 << 20));   // D*D    f16 = 2 MB
    _Float16* Wkt = (_Float16*)(ws + ((size_t)18 << 20));
    _Float16* Wvt = (_Float16*)(ws + ((size_t)20 << 20));
    _Float16* qh  = (_Float16*)(ws + ((size_t)22 << 20));   // (B,H,T,64) 8 MB
    _Float16* kh  = (_Float16*)(ws + ((size_t)30 << 20));   // (B,H,N,64) 8 MB
    _Float16* vth = (_Float16*)(ws + ((size_t)38 << 20));   // (B,H,64,N) 8 MB

    // 1) layernorms -> f16
    ln_f16_kernel<<<Bsz * Tq, 256, 0, stream>>>(x,  gamma_x, beta_x, xn, Dd);
    ln_f16_kernel<<<Bsz * Nk, 256, 0, stream>>>(xf, gamma_f, beta_f, fn, Ll);

    // 2) transpose + convert weights
    const int welems = Dd * Ll;
    wt_f16_kernel<<<(welems + 255) / 256, 256, 0, stream>>>(Wq, Wqt, Dd, Dd);
    wt_f16_kernel<<<(welems + 255) / 256, 256, 0, stream>>>(Wk, Wkt, Ll, Dd);
    wt_f16_kernel<<<(welems + 255) / 256, 256, 0, stream>>>(Wv, Wvt, Ll, Dd);

    // 3) projections (M=4096, K=1024, Ncol=1024):
    //    wave-tiles = 256 * 16 = 4096 (16 rows x 64 cols each), 4 waves/block
    const int gemmBlocks = (256 * 16) / 4;
    gemm_wmma_kernel<<<gemmBlocks, 128, 0, stream>>>(xn, Wqt, qh,  Bsz * Tq, Dd, Dd, 0, Tq);
    gemm_wmma_kernel<<<gemmBlocks, 128, 0, stream>>>(fn, Wkt, kh,  Bsz * Nk, Ll, Dd, 0, Nk);
    gemm_wmma_kernel<<<gemmBlocks, 128, 0, stream>>>(fn, Wvt, vth, Bsz * Nk, Ll, Dd, 1, Nk);

    // 4) attention + residual: B*H*(T/16) = 4096 wave-tiles, 4 waves/block
    attn_wmma_kernel<<<(Bsz * Hh * (Tq / 16)) / 4, 128, 0, stream>>>(qh, kh, vth, x, out);
}
